// HeARBiMambaHybridClassifier_16149077033375
// MI455X (gfx1250) — matrix operand order repaired and checked by reading, not compiled, not run on hardware
//
#include <hip/hip_runtime.h>
#include <math.h>

// ---------------- model constants ----------------
#define DMODEL 1024
#define DINNER 2048
#define DSTATE 16
#define DTRANK 64
#define NHEADS 8
#define DHEAD  128
#define BB     8
#define LL     512
#define XPROJ  (DTRANK + 2*DSTATE)   // 96

typedef __bf16 bf16x16 __attribute__((ext_vector_type(16)));
typedef float  f32x8   __attribute__((ext_vector_type(8)));

__device__ __forceinline__ unsigned short f2bf(float f) {
  unsigned int u = __builtin_bit_cast(unsigned int, f);
  u = u + 0x7fffu + ((u >> 16) & 1u);          // round-to-nearest-even
  return (unsigned short)(u >> 16);
}

// ---------------- bf16 WMMA GEMM: C = act(A @ W^T [+bias]) [+C] ----------------
// A: (M,K) bf16 row-major, lda mult of 8.  W: (N,K) bf16 row-major, ldw mult of 8.
// C: (M,N) f32.  Batched via blockIdx.y with explicit strides.
// Register-blocked: one 64x32 macro-tile (4x2 WMMA tiles) per wave,
// 8 waves per block, K stepped by 32.  M must be a multiple of 64, N of 32.
// Per K-step: 4 A-fragments + 2 B-fragments feed 8 independent v_wmma chains
// (0.75 fragment loads per wmma vs 2.0 for unblocked -> 2.7x less L2 traffic).
__global__ __launch_bounds__(256) void k_wmma_gemm(
    const unsigned short* __restrict__ A, int lda, long sA,
    const unsigned short* __restrict__ W, int ldw, long sW,
    float* __restrict__ C, int ldc, long sC,
    const float* __restrict__ bias,
    int M, int N, int K, int act, int acc)
{
  const int wave = threadIdx.x >> 5;
  const int lane = threadIdx.x & 31;
  const int tilesN = N >> 5;            // 32-wide macro columns
  const int tilesM = M >> 6;            // 64-tall macro rows
  int tile = blockIdx.x * 8 + wave;
  if (tile >= tilesM * tilesN) return;
  const int tm = tile / tilesN;
  const int tn = tile - tm * tilesN;
  const long bat = blockIdx.y;
  const unsigned short* Ab = A + bat * sA;
  const unsigned short* Wb = W + bat * sW;
  float* Cb = C + bat * sC;

  const int half16 = lane >> 4;   // 0 or 1 (lane half)
  const int l15    = lane & 15;
  // A fragment rows: lane half selects K {0..7,16..23} vs {8..15,24..31}
  const unsigned short* aptr = Ab + (long)(tm*64 + l15) * lda + half16 * 8;
  const long aRowStep = (long)16 * lda;          // next 16-row A tile
  // B fragment rows: lane half selects K {0..7,16..23} vs {8..15,24..31}
  const unsigned short* wptr = Wb + (long)(tn*32 + l15) * ldw + half16 * 8;
  const long wColStep = (long)16 * ldw;          // next 16-col B tile

  union Frag { bf16x16 v; uint4 q[2]; };
  f32x8 c[4][2] = {};

  for (int k0 = 0; k0 < K; k0 += 32) {
    Frag a[4], w[2];
    #pragma unroll
    for (int r = 0; r < 4; ++r) {
      const unsigned short* p = aptr + r * aRowStep + k0;
      a[r].q[0] = *reinterpret_cast<const uint4*>(p);        // K = k0+h8    .. +7
      a[r].q[1] = *reinterpret_cast<const uint4*>(p + 16);   // K = k0+16+h8 .. +7
    }
    #pragma unroll
    for (int n = 0; n < 2; ++n) {
      const unsigned short* p = wptr + n * wColStep + k0;
      w[n].q[0] = *reinterpret_cast<const uint4*>(p);        // K = k0+h8    .. +7
      w[n].q[1] = *reinterpret_cast<const uint4*>(p + 16);   // K = k0+16+h8 .. +7
    }
    #pragma unroll
    for (int r = 0; r < 4; ++r)
      #pragma unroll
      for (int n = 0; n < 2; ++n)
        c[r][n] = __builtin_amdgcn_wmma_f32_16x16x32_bf16(
            false, a[r].v, false, w[n].v, (short)0, c[r][n], false, false);
  }

  #pragma unroll
  for (int r = 0; r < 4; ++r) {
    const int row0 = tm*64 + r*16 + half16*8;   // VGPR v -> row0+v
    #pragma unroll
    for (int n = 0; n < 2; ++n) {
      const int col = tn*32 + n*16 + l15;
      const float bv = bias ? bias[col] : 0.0f;
      #pragma unroll
      for (int v = 0; v < 8; ++v) {
        float x = c[r][n][v] + bv;
        if (act == 1) x = (x > 20.0f) ? x : log1pf(__expf(x));   // softplus
        float* p = Cb + (long)(row0 + v) * ldc + col;
        if (acc) x += *p;
        *p = x;
      }
    }
  }
}

// ---------------- f32 -> bf16 (optionally strided sub-matrix) ----------------
__global__ __launch_bounds__(256) void k_to_bf16(
    const float* __restrict__ src, unsigned short* __restrict__ dst,
    long rows, int cols, int ld)
{
  long idx = (long)blockIdx.x * 256 + threadIdx.x;
  if (idx >= rows * (long)cols) return;
  long r = idx / cols; int cc = (int)(idx - r * cols);
  dst[idx] = f2bf(src[r * (long)ld + cc]);
}

// ---------------- depthwise causal/anticausal conv + bias + SiLU ----------------
__global__ __launch_bounds__(256) void k_conv_silu(
    const float* __restrict__ xz, const float* __restrict__ cw,
    const float* __restrict__ cb, float* __restrict__ xc,
    unsigned short* __restrict__ xc_bf, int dir)
{
  long idx = (long)blockIdx.x * 256 + threadIdx.x;
  if (idx >= (long)BB * LL * DINNER) return;
  int d = (int)(idx % DINNER);
  long bl = idx / DINNER;
  int t = (int)(bl % LL);
  long brow = bl - t;                       // b*L
  float s = cb[d];
  if (dir == 0) {                           // causal: sum_k w[k]*x[t-3+k]
    #pragma unroll
    for (int k = 0; k < 4; ++k) {
      int tt = t - 3 + k;
      if (tt >= 0) s += cw[d*4 + k] * xz[(brow + tt) * (long)(2*DINNER) + d];
    }
  } else {                                  // reversed stream: sum_j w[3-j]*x[t+j]
    #pragma unroll
    for (int j = 0; j < 4; ++j) {
      int tt = t + j;
      if (tt < LL) s += cw[d*4 + 3 - j] * xz[(brow + tt) * (long)(2*DINNER) + d];
    }
  }
  float v = s / (1.0f + __expf(-s));        // SiLU
  xc[idx] = v;
  xc_bf[idx] = f2bf(v);
}

// ---------------- selective scan: one thread per (b,d) channel ----------------
// Includes +xc*D and *silu(z) epilogue; emits bf16 y for out_proj.
__global__ __launch_bounds__(256) void k_scan(
    const float* __restrict__ dt, const float* __restrict__ xdbl,
    const float* __restrict__ xc, const float* __restrict__ xz,
    const float* __restrict__ A_log, const float* __restrict__ Dp,
    unsigned short* __restrict__ y_bf, int dir)
{
  int g = blockIdx.x * 256 + threadIdx.x;
  if (g >= BB * DINNER) return;
  int d = g % DINNER;
  int b = g / DINNER;
  float a[DSTATE], h[DSTATE];
  #pragma unroll
  for (int n = 0; n < DSTATE; ++n) { a[n] = -__expf(A_log[d*DSTATE + n]); h[n] = 0.0f; }
  float Dv = Dp[d];
  for (int s = 0; s < LL; ++s) {
    int t = dir ? (LL - 1 - s) : s;
    long row = (long)b * LL + t;
    float dtv = dt[row * DINNER + d];
    float xv  = xc[row * DINNER + d];
    const float* bc = xdbl + row * XPROJ;
    float dx = dtv * xv;
    float acc = 0.0f;
    #pragma unroll
    for (int n = 0; n < DSTATE; ++n) {
      h[n] = __expf(dtv * a[n]) * h[n] + dx * bc[DTRANK + n];
      acc += h[n] * bc[DTRANK + DSTATE + n];
    }
    float yv = acc + xv * Dv;
    float z = xz[row * (long)(2*DINNER) + DINNER + d];
    yv *= z / (1.0f + __expf(-z));
    y_bf[row * DINNER + d] = f2bf(yv);
  }
}

// ---------------- QKV bias-add + head split; V stored transposed (dh,L) ----------------
__global__ __launch_bounds__(256) void k_qkv_repack(
    const float* __restrict__ qkv, const float* __restrict__ bias,
    unsigned short* __restrict__ Q, unsigned short* __restrict__ Kb,
    unsigned short* __restrict__ Vt)
{
  long idx = (long)blockIdx.x * 256 + threadIdx.x;
  if (idx >= (long)BB * LL * DMODEL) return;
  int dmd = (int)(idx % DMODEL);
  long bl = idx / DMODEL;
  int l = (int)(bl % LL);
  int b = (int)(bl / LL);
  int hh = dmd / DHEAD, dd = dmd % DHEAD;
  const float* base = qkv + bl * (long)(3*DMODEL);
  float q = base[dmd]            + bias[dmd];
  float k = base[DMODEL + dmd]   + bias[DMODEL + dmd];
  float v = base[2*DMODEL + dmd] + bias[2*DMODEL + dmd];
  long bh = (long)b * NHEADS + hh;
  Q [(bh*LL + l)*DHEAD + dd] = f2bf(q);
  Kb[(bh*LL + l)*DHEAD + dd] = f2bf(k);
  Vt[(bh*DHEAD + dd)*LL + l] = f2bf(v);
}

// ---------------- scaled softmax over rows of 512; bf16 probabilities out ----------------
__global__ __launch_bounds__(256) void k_softmax(
    const float* __restrict__ sc, unsigned short* __restrict__ att)
{
  long row = blockIdx.x;
  int tid = threadIdx.x;
  const float scale = 0.088388347648318447f;  // 1/sqrt(128)
  const float* r = sc + row * (long)LL;
  float a0 = r[tid] * scale;
  float a1 = r[tid + 256] * scale;
  __shared__ float red[256];
  red[tid] = fmaxf(a0, a1);
  __syncthreads();
  for (int st = 128; st > 0; st >>= 1) { if (tid < st) red[tid] = fmaxf(red[tid], red[tid+st]); __syncthreads(); }
  float m = red[0];
  __syncthreads();
  float e0 = __expf(a0 - m), e1 = __expf(a1 - m);
  red[tid] = e0 + e1;
  __syncthreads();
  for (int st = 128; st > 0; st >>= 1) { if (tid < st) red[tid] += red[tid+st]; __syncthreads(); }
  float inv = 1.0f / red[0];
  att[row*(long)LL + tid]       = f2bf(e0 * inv);
  att[row*(long)LL + tid + 256] = f2bf(e1 * inv);
}

// ---------------- (B,H,L,dh) f32 -> (B,L,E) bf16 ----------------
__global__ __launch_bounds__(256) void k_o_perm(
    const float* __restrict__ o, unsigned short* __restrict__ obf)
{
  long idx = (long)blockIdx.x * 256 + threadIdx.x;
  if (idx >= (long)BB * LL * DMODEL) return;
  int dd = (int)(idx % DHEAD);
  long r = idx / DHEAD;
  int l = (int)(r % LL); r /= LL;
  int hh = (int)(r % NHEADS);
  int b = (int)(r / NHEADS);
  obf[((long)(b*LL + l)) * DMODEL + hh*DHEAD + dd] = f2bf(o[idx]);
}

// ---------------- LayerNorm over last dim (1024), one block per row ----------------
__global__ __launch_bounds__(256) void k_ln(
    const float* __restrict__ in, const float* __restrict__ g,
    const float* __restrict__ bta, float* __restrict__ out)
{
  long row = blockIdx.x;
  int tid = threadIdx.x;
  const float* r = in + row * (long)DMODEL;
  float v[4]; float s = 0.0f;
  #pragma unroll
  for (int i = 0; i < 4; ++i) { v[i] = r[tid + i*256]; s += v[i]; }
  __shared__ float red[256];
  red[tid] = s; __syncthreads();
  for (int st = 128; st > 0; st >>= 1) { if (tid < st) red[tid] += red[tid+st]; __syncthreads(); }
  float mean = red[0] * (1.0f / DMODEL);
  __syncthreads();
  float vs = 0.0f;
  #pragma unroll
  for (int i = 0; i < 4; ++i) { float d = v[i] - mean; vs += d * d; }
  red[tid] = vs; __syncthreads();
  for (int st = 128; st > 0; st >>= 1) { if (tid < st) red[tid] += red[tid+st]; __syncthreads(); }
  float rstd = rsqrtf(red[0] * (1.0f / DMODEL) + 1e-5f);
  #pragma unroll
  for (int i = 0; i < 4; ++i) {
    int cc = tid + i*256;
    out[row*(long)DMODEL + cc] = (v[i] - mean) * rstd * g[cc] + bta[cc];
  }
}

// ---------------- mean+max pool over L ----------------
__global__ __launch_bounds__(256) void k_pool(
    const float* __restrict__ xh, float* __restrict__ comb)
{
  int idx = blockIdx.x * 256 + threadIdx.x;
  if (idx >= BB * DMODEL) return;
  int d = idx % DMODEL, b = idx / DMODEL;
  float s = 0.0f, mx = -3.402823466e38f;
  for (int t = 0; t < LL; ++t) {
    float x = xh[((long)b*LL + t) * DMODEL + d];
    s += x; mx = fmaxf(mx, x);
  }
  comb[idx] = s * (1.0f / LL) + mx;
}

// ---------------- classifier head: fc512 + BN + exact GELU + fc1 ----------------
__global__ __launch_bounds__(512) void k_clf(
    const float* __restrict__ comb, const float* __restrict__ w1,
    const float* __restrict__ b1, const float* __restrict__ bng,
    const float* __restrict__ bnb, const float* __restrict__ bnm,
    const float* __restrict__ bnv, const float* __restrict__ w2,
    const float* __restrict__ b2, float* __restrict__ out)
{
  int b = blockIdx.x;
  int j = threadIdx.x;   // 512
  const float* cr = comb + b * DMODEL;
  const float* wr = w1 + (long)j * DMODEL;
  float s = 0.0f;
  for (int k = 0; k < DMODEL; ++k) s += cr[k] * wr[k];
  s += b1[j];
  s = (s - bnm[j]) * rsqrtf(bnv[j] + 1e-5f) * bng[j] + bnb[j];
  float hgelu = 0.5f * s * (1.0f + erff(s * 0.70710678118654752f));
  __shared__ float red[512];
  red[j] = hgelu * w2[j];
  __syncthreads();
  for (int st = 256; st > 0; st >>= 1) { if (j < st) red[j] += red[j+st]; __syncthreads(); }
  if (j == 0) out[b] = red[0] + b2[0];
}

// ======================= host orchestration =======================
extern "C" void kernel_launch(void* const* d_in, const int* in_sizes, int n_in,
                              void* d_out, int out_size, void* d_ws, size_t ws_size,
                              hipStream_t stream)
{
  (void)in_sizes; (void)n_in; (void)out_size; (void)ws_size;
  const float* x = (const float*)d_in[0];
  const float* fw[9]; const float* bw[9];
  for (int i = 0; i < 9; ++i) { fw[i] = (const float*)d_in[1+i]; bw[i] = (const float*)d_in[10+i]; }
  // param order: 0 in_proj_w, 1 conv_w, 2 conv_b, 3 x_proj_w, 4 dt_proj_w,
  //              5 dt_proj_b, 6 A_log, 7 D, 8 out_proj_w
  const float* attn_in_w  = (const float*)d_in[19];
  const float* attn_in_b  = (const float*)d_in[20];
  const float* attn_out_w = (const float*)d_in[21];
  const float* attn_out_b = (const float*)d_in[22];
  const float* attn_g   = (const float*)d_in[23];
  const float* attn_bta = (const float*)d_in[24];
  const float* norm_g   = (const float*)d_in[25];
  const float* norm_bta = (const float*)d_in[26];
  const float* w1  = (const float*)d_in[27];
  const float* b1  = (const float*)d_in[28];
  const float* bng = (const float*)d_in[29];
  const float* bnb = (const float*)d_in[30];
  const float* bnm = (const float*)d_in[31];
  const float* bnv = (const float*)d_in[32];
  const float* w2  = (const float*)d_in[33];
  const float* b2  = (const float*)d_in[34];

  char* ws = (char*)d_ws;
  size_t off = 0;
  auto alloc = [&](size_t bytes) -> char* {
    off = (off + 255) & ~(size_t)255;
    char* p = ws + off;
    off += bytes;
    return p;
  };
  const long NT = (long)BB * LL;   // 4096 tokens

  // ---- persistent arena ----
  unsigned short* w_in   = (unsigned short*)alloc(2ull * 2*DINNER * DMODEL);
  unsigned short* w_xp   = (unsigned short*)alloc(2ull * XPROJ * DINNER);
  unsigned short* w_dt   = (unsigned short*)alloc(2ull * DINNER * DTRANK);
  unsigned short* w_out  = (unsigned short*)alloc(2ull * DMODEL * DINNER);
  unsigned short* w_ain  = (unsigned short*)alloc(2ull * 3*DMODEL * DMODEL);
  unsigned short* w_aout = (unsigned short*)alloc(2ull * DMODEL * DMODEL);
  unsigned short* x_bf   = (unsigned short*)alloc(2ull * NT * DMODEL);
  float* xm = (float*)alloc(4ull * NT * DMODEL);
  size_t scratch0 = off;

  // ---- mamba arena (aliased, phases are sequential) ----
  float* xz = (float*)alloc(4ull * NT * 2*DINNER);
  float* xc = (float*)alloc(4ull * NT * DINNER);
  unsigned short* xc_bf = (unsigned short*)alloc(2ull * NT * DINNER);
  float* xdbl = (float*)alloc(4ull * NT * XPROJ);
  unsigned short* dtin_bf = (unsigned short*)alloc(2ull * NT * DTRANK);
  float* dtb = (float*)alloc(4ull * NT * DINNER);
  unsigned short* y_bf = (unsigned short*)alloc(2ull * NT * DINNER);

  // ---- MHA arena (aliases mamba arena) ----
  off = scratch0;
  unsigned short* xm_bf = (unsigned short*)alloc(2ull * NT * DMODEL);
  float* qkv = (float*)alloc(4ull * NT * 3*DMODEL);
  unsigned short* Qbf = (unsigned short*)alloc(2ull * NT * DMODEL);
  unsigned short* Kbf = (unsigned short*)alloc(2ull * NT * DMODEL);
  unsigned short* Vtbf = (unsigned short*)alloc(2ull * NT * DMODEL);
  float* scores = (float*)alloc(4ull * BB * NHEADS * LL * LL);
  unsigned short* att_bf = (unsigned short*)alloc(2ull * BB * NHEADS * LL * LL);
  float* o = (float*)alloc(4ull * NT * DMODEL);
  unsigned short* o_bf = (unsigned short*)alloc(2ull * NT * DMODEL);

  // ---- post arena (aliases both, used after MHA) ----
  off = scratch0;
  float* xh1  = (float*)alloc(4ull * NT * DMODEL);
  float* xh2  = (float*)alloc(4ull * NT * DMODEL);
  float* comb = (float*)alloc(4ull * BB * DMODEL);

  auto cvt = [&](const float* src, unsigned short* dst, long rows, int cols, int ld) {
    long n = rows * (long)cols;
    k_to_bf16<<<dim3((unsigned)((n + 255) / 256)), dim3(256), 0, stream>>>(src, dst, rows, cols, ld);
  };
  auto gemm = [&](const unsigned short* A, int lda, long sA,
                  const unsigned short* W, int ldw, long sW,
                  float* C, int ldc, long sC, const float* bias,
                  int M, int N, int K, int nbat, int act, int acc) {
    int tiles = (M/64) * (N/32);         // 64x32 macro-tiles, one per wave
    dim3 grid((unsigned)((tiles + 7) / 8), (unsigned)nbat);
    k_wmma_gemm<<<grid, dim3(256), 0, stream>>>(A, lda, sA, W, ldw, sW,
                                                C, ldc, sC, bias, M, N, K, act, acc);
  };

  // ---- shared input / attention weight conversion ----
  cvt(x, x_bf, NT, DMODEL, DMODEL);
  cvt(attn_in_w,  w_ain,  3*DMODEL, DMODEL, DMODEL);
  cvt(attn_out_w, w_aout, DMODEL,   DMODEL, DMODEL);

  // ---- bidirectional mamba: token-wise GEMMs direction-invariant;
  //      only conv + scan take the direction flag ----
  for (int pass = 0; pass < 2; ++pass) {
    const float* const* P = pass ? bw : fw;
    int dir = pass;
    cvt(P[0], w_in,  2*DINNER, DMODEL, DMODEL);
    cvt(P[3], w_xp,  XPROJ,    DINNER, DINNER);
    cvt(P[4], w_dt,  DINNER,   DTRANK, DTRANK);
    cvt(P[8], w_out, DMODEL,   DINNER, DINNER);

    // xz = x @ in_proj_w^T   (4096 x 4096 x 1024)
    gemm(x_bf, DMODEL, 0, w_in, DMODEL, 0, xz, 2*DINNER, 0, nullptr,
         (int)NT, 2*DINNER, DMODEL, 1, 0, 0);
    // depthwise conv + bias + SiLU
    k_conv_silu<<<dim3((unsigned)((NT*DINNER + 255)/256)), 256, 0, stream>>>(
        xz, P[1], P[2], xc, xc_bf, dir);
    // x_dbl = xc @ x_proj_w^T   (4096 x 96 x 2048)
    gemm(xc_bf, DINNER, 0, w_xp, DINNER, 0, xdbl, XPROJ, 0, nullptr,
         (int)NT, XPROJ, DINNER, 1, 0, 0);
    // dt input slice -> bf16
    cvt(xdbl, dtin_bf, NT, DTRANK, XPROJ);
    // dt = softplus(dt_in @ dt_proj_w^T + b)   (4096 x 2048 x 64)
    gemm(dtin_bf, DTRANK, 0, w_dt, DTRANK, 0, dtb, DINNER, 0, P[5],
         (int)NT, DINNER, DTRANK, 1, 1, 0);
    // selective scan (+ D skip + silu(z) gate) -> y_bf
    k_scan<<<dim3((BB*DINNER + 255)/256), 256, 0, stream>>>(
        dtb, xdbl, xc, xz, P[6], P[7], y_bf, dir);
    // xm (+)= y @ out_proj_w^T   (4096 x 1024 x 2048); fwd writes, bwd accumulates
    gemm(y_bf, DINNER, 0, w_out, DINNER, 0, xm, DMODEL, 0, nullptr,
         (int)NT, DMODEL, DINNER, 1, 0, pass);
  }

  // ---- MHA ----
  cvt(xm, xm_bf, NT, DMODEL, DMODEL);
  // qkv = xm @ in_proj_w^T   (4096 x 3072 x 1024), bias added in repack
  gemm(xm_bf, DMODEL, 0, w_ain, DMODEL, 0, qkv, 3*DMODEL, 0, nullptr,
       (int)NT, 3*DMODEL, DMODEL, 1, 0, 0);
  k_qkv_repack<<<dim3((unsigned)((NT*DMODEL + 255)/256)), 256, 0, stream>>>(
      qkv, attn_in_b, Qbf, Kbf, Vtbf);
  // scores = Q @ K^T per (b,h): 64 batches of 512 x 512 x 128
  gemm(Qbf, DHEAD, (long)LL*DHEAD, Kbf, DHEAD, (long)LL*DHEAD,
       scores, LL, (long)LL*LL, nullptr, LL, LL, DHEAD, BB*NHEADS, 0, 0);
  k_softmax<<<dim3((unsigned)(BB*NHEADS*LL)), 256, 0, stream>>>(scores, att_bf);
  // o = att @ V per (b,h): 64 batches of 512 x 128 x 512 (V stored transposed)
  gemm(att_bf, LL, (long)LL*LL, Vtbf, LL, (long)DHEAD*LL,
       o, DHEAD, (long)LL*DHEAD, nullptr, LL, DHEAD, LL, BB*NHEADS, 0, 0);
  k_o_perm<<<dim3((unsigned)((NT*DMODEL + 255)/256)), 256, 0, stream>>>(o, o_bf);
  // xm += o @ out_w^T + out_b   (4096 x 1024 x 1024)
  gemm(o_bf, DMODEL, 0, w_aout, DMODEL, 0, xm, DMODEL, 0, attn_out_b,
       (int)NT, DMODEL, DMODEL, 1, 0, 1);

  // ---- LN x2, pool, classifier ----
  k_ln<<<dim3((unsigned)NT), 256, 0, stream>>>(xm,  attn_g, attn_bta, xh1);
  k_ln<<<dim3((unsigned)NT), 256, 0, stream>>>(xh1, norm_g, norm_bta, xh2);
  k_pool<<<dim3((BB*DMODEL + 255)/256), 256, 0, stream>>>(xh2, comb);
  k_clf<<<dim3(BB), dim3(512), 0, stream>>>(comb, w1, b1, bng, bnb, bnm, bnv,
                                            w2, b2, (float*)d_out);
}
